// PackedLinear_12867722019333
// MI455X (gfx1250) — 8-bit matrix operand order repaired and checked by reading, compiled, not run on hardware
//
#include <hip/hip_runtime.h>
#include <stdint.h>


typedef int          v8i __attribute__((ext_vector_type(8)));
typedef unsigned int v2u __attribute__((ext_vector_type(2)));
typedef unsigned int v4u __attribute__((ext_vector_type(4)));

#define M_DIM 8192
#define K_DIM 4096
#define N_DIM 4096

__global__ __launch_bounds__(256) void cvt_x_kernel(const float* __restrict__ x,
                                                    uint8_t* __restrict__ A8,
                                                    int n8) {
  int t = blockIdx.x * 256 + threadIdx.x;
  if (t >= n8) return;
  const float4* xp = (const float4*)x + (size_t)t * 2;
  float4 f0 = xp[0];
  float4 f1 = xp[1];
  unsigned b0 = (unsigned)f0.x | ((unsigned)f0.y << 8) |
                ((unsigned)f0.z << 16) | ((unsigned)f0.w << 24);
  unsigned b1 = (unsigned)f1.x | ((unsigned)f1.y << 8) |
                ((unsigned)f1.z << 16) | ((unsigned)f1.w << 24);
  v2u o;
  o.x = b0;
  o.y = b1;
  *((v2u*)A8 + t) = o;
}

__global__ __launch_bounds__(256) void cvt_w_kernel(const int* __restrict__ wp,
                                                    uint8_t* __restrict__ W8,
                                                    int n4) {
  int t = blockIdx.x * 256 + threadIdx.x;
  if (t >= n4) return;
  int4 w = ((const int4*)wp)[t];
  unsigned p0 = (unsigned)(w.x & 0xF) | ((unsigned)((w.x >> 4) & 0xF) << 8) |
                ((unsigned)(w.y & 0xF) << 16) | ((unsigned)((w.y >> 4) & 0xF) << 24);
  unsigned p1 = (unsigned)(w.z & 0xF) | ((unsigned)((w.z >> 4) & 0xF) << 8) |
                ((unsigned)(w.w & 0xF) << 16) | ((unsigned)((w.w >> 4) & 0xF) << 24);
  v2u o;
  o.x = p0;
  o.y = p1;
  *((v2u*)W8 + t) = o;
}


struct Frags {
  v8i a0, a1;
  v8i b0, b1, b2, b3;
};

__device__ __forceinline__ void load_frags(Frags& f,
                                           const uint8_t* __restrict__ arow0,
                                           const uint8_t* __restrict__ arow1,
                                           const uint8_t* __restrict__ brow0,
                                           int k, int aoff, int boff) {
  union { v8i v; v2u h[4]; } a0u, a1u;
  a0u.h[0] = *(const v2u*)(arow0 + k + aoff);
  a0u.h[1] = *(const v2u*)(arow0 + k + 8 + aoff);
  a0u.h[2] = *(const v2u*)(arow0 + k + 32 + aoff);
  a0u.h[3] = *(const v2u*)(arow0 + k + 40 + aoff);
  a1u.h[0] = *(const v2u*)(arow1 + k + aoff);
  a1u.h[1] = *(const v2u*)(arow1 + k + 8 + aoff);
  a1u.h[2] = *(const v2u*)(arow1 + k + 32 + aoff);
  a1u.h[3] = *(const v2u*)(arow1 + k + 40 + aoff);
  f.a0 = a0u.v;
  f.a1 = a1u.v;

  union { v8i v; v4u q[2]; } bu0, bu1, bu2, bu3;
  const uint8_t* b0p = brow0 + k + boff;
  bu0.q[0] = *(const v4u*)(b0p);
  bu0.q[1] = *(const v4u*)(b0p + 32);
  bu1.q[0] = *(const v4u*)(b0p + (size_t)16 * K_DIM);
  bu1.q[1] = *(const v4u*)(b0p + (size_t)16 * K_DIM + 32);
  bu2.q[0] = *(const v4u*)(b0p + (size_t)32 * K_DIM);
  bu2.q[1] = *(const v4u*)(b0p + (size_t)32 * K_DIM + 32);
  bu3.q[0] = *(const v4u*)(b0p + (size_t)48 * K_DIM);
  bu3.q[1] = *(const v4u*)(b0p + (size_t)48 * K_DIM + 32);
  f.b0 = bu0.v;
  f.b1 = bu1.v;
  f.b2 = bu2.v;
  f.b3 = bu3.v;
}

__device__ __forceinline__ void mma_frags(const Frags& f, v8i acc[2][4]) {
  acc[0][0] = __builtin_amdgcn_wmma_i32_16x16x64_iu8(false, f.a0, false, f.b0,
                                                     acc[0][0], false, false);
  acc[1][0] = __builtin_amdgcn_wmma_i32_16x16x64_iu8(false, f.a1, false, f.b0,
                                                     acc[1][0], false, false);
  acc[0][1] = __builtin_amdgcn_wmma_i32_16x16x64_iu8(false, f.a0, false, f.b1,
                                                     acc[0][1], false, false);
  acc[1][1] = __builtin_amdgcn_wmma_i32_16x16x64_iu8(false, f.a1, false, f.b1,
                                                     acc[1][1], false, false);
  acc[0][2] = __builtin_amdgcn_wmma_i32_16x16x64_iu8(false, f.a0, false, f.b2,
                                                     acc[0][2], false, false);
  acc[1][2] = __builtin_amdgcn_wmma_i32_16x16x64_iu8(false, f.a1, false, f.b2,
                                                     acc[1][2], false, false);
  acc[0][3] = __builtin_amdgcn_wmma_i32_16x16x64_iu8(false, f.a0, false, f.b3,
                                                     acc[0][3], false, false);
  acc[1][3] = __builtin_amdgcn_wmma_i32_16x16x64_iu8(false, f.a1, false, f.b3,
                                                     acc[1][3], false, false);
}

__global__ __launch_bounds__(256) void packed_linear_wmma_kernel(
    const uint8_t* __restrict__ A8, const uint8_t* __restrict__ W8,
    const float* __restrict__ bias, float* __restrict__ out) {
  const int lane   = threadIdx.x & 31;
  const int wave   = threadIdx.x >> 5;
  const int tileId = blockIdx.x * 8 + wave;
  const int nTiles = N_DIM / 64;
  const int mTile  = tileId / nTiles;
  const int nTile  = tileId % nTiles;
  const int mBase  = mTile * 32;
  const int nBase  = nTile * 64;
  const int l16    = lane & 15;
  const int half   = lane >> 4;

  const uint8_t* arow0 = A8 + (size_t)(mBase + l16) * K_DIM;
  const uint8_t* arow1 = arow0 + (size_t)16 * K_DIM;
  const uint8_t* brow0 = W8 + (size_t)(nBase + l16) * K_DIM;

  v8i acc[2][4];
#pragma unroll
  for (int m = 0; m < 2; ++m)
#pragma unroll
    for (int j = 0; j < 4; ++j)
      acc[m][j] = (v8i)0;

  const int aoff = half * 16;
  const int boff = half * 16;

  Frags f0, f1;
  load_frags(f0, arow0, arow1, brow0, 0, aoff, boff);
  load_frags(f1, arow0, arow1, brow0, 64, aoff, boff);

#pragma unroll 1
  for (int k = 0; k < K_DIM - 128; k += 128) {
    mma_frags(f0, acc);
    load_frags(f0, arow0, arow1, brow0, k + 128, aoff, boff);
    mma_frags(f1, acc);
    load_frags(f1, arow0, arow1, brow0, k + 192, aoff, boff);
  }
  mma_frags(f0, acc);
  mma_frags(f1, acc);

#pragma unroll
  for (int j = 0; j < 4; ++j) {
    const int col  = nBase + j * 16 + l16;
    const float bv = bias[col];
#pragma unroll
    for (int m = 0; m < 2; ++m) {
      const int rbase = mBase + m * 16 + half * 8;
      float* o = out + (size_t)rbase * N_DIM + col;
#pragma unroll
      for (int r = 0; r < 8; ++r)
        o[(size_t)r * N_DIM] = (float)acc[m][j][r] + bv;
    }
  }
}

extern "C" void kernel_launch(void* const* d_in, const int* in_sizes, int n_in,
                              void* d_out, int out_size, void* d_ws, size_t ws_size,
                              hipStream_t stream) {
  const float* x    = (const float*)d_in[0];
  const int*   wp   = (const int*)d_in[1];
  const float* bias = (const float*)d_in[2];
  float*       out  = (float*)d_out;

  uint8_t* A8 = (uint8_t*)d_ws;
  uint8_t* W8 = A8 + (size_t)M_DIM * K_DIM;

  const int n8 = M_DIM * K_DIM / 8;
  cvt_x_kernel<<<n8 / 256, 256, 0, stream>>>(x, A8, n8);

  const int n4 = N_DIM * (K_DIM / 2) / 4;
  cvt_w_kernel<<<n4 / 256, 256, 0, stream>>>(wp, W8, n4);

  const int tiles  = (M_DIM / 32) * (N_DIM / 64);
  const int blocks = tiles / 8;
  packed_linear_wmma_kernel<<<blocks, 256, 0, stream>>>(A8, W8, bias, out);
}
